// GAT_71854802862590
// MI455X (gfx1250) — matrix operand order repaired and checked by reading, not compiled, not run on hardware
//
#include <hip/hip_runtime.h>
#include <hip/hip_bf16.h>
#include <math.h>

typedef __attribute__((ext_vector_type(16))) __bf16 v16bf;
typedef __attribute__((ext_vector_type(8)))  float  v8f;

#define LRELU 0.2f
#define EPSD  1e-16f

// ---- ordered-uint encoding so atomicMax(uint) == max(float) ----------------
__device__ __forceinline__ unsigned f2ord(float f) {
  unsigned b = __float_as_uint(f);
  return (b & 0x80000000u) ? ~b : (b | 0x80000000u);
}
__device__ __forceinline__ float ord2f(unsigned k) {
  return __uint_as_float((k & 0x80000000u) ? (k ^ 0x80000000u) : ~k);
}
#define ORD_NEG_INF 0x007FFFFFu  // f2ord(-inf)

// ---- generic fill (graph-capture-safe replacement for hipMemsetAsync) -----
__global__ void fill_u32(unsigned* __restrict__ p, unsigned v, size_t n) {
  size_t i = (size_t)blockIdx.x * blockDim.x + threadIdx.x;
  size_t s = (size_t)gridDim.x * blockDim.x;
  for (; i < n; i += s) p[i] = v;
}

// ---- WMMA GEMM: D[M x ncol] = A[M x K] * B[K x ncol], f32 via bf16 hi/lo ---
// 256 threads = 8 waves; waves tile (8/NCT) row-tiles x NCT col-tiles.
// B is staged into LDS in WMMA-fragment order so every lane's 16 bf16 are
// contiguous -> ds_load_b128 pairs (no scalar ds_load_u16 gathers).
template <int NCT>
__global__ __launch_bounds__(256)
void gemm_bf16x(const float* __restrict__ A, const float* __restrict__ B,
                float* __restrict__ D, int M, int K, int ncol) {
  constexpr int NCOLPAD = NCT * 16;
  constexpr int ROWS_PB = (8 / NCT) * 16;  // 16 (NCT=8) or 64 (NCT=2)
  __shared__ __bf16 As_hi[64 * 32], As_lo[64 * 32];
  __shared__ __bf16 Bs_hi[8 * 512], Bs_lo[8 * 512];  // per col-tile: 32 lanes x 16

  const int tid  = threadIdx.x;
  const int wave = tid >> 5;
  const int lane = tid & 31;
  const int col_tile = wave % NCT;
  const int row_tile = wave / NCT;
  const int row0 = blockIdx.x * ROWS_PB;

  const int mlow = lane & 15;
  const int half = lane >> 4;
  v8f acc = {};

  const __bf16* Ah = &As_hi[(row_tile * 16 + mlow) * 32];
  const __bf16* Al = &As_lo[(row_tile * 16 + mlow) * 32];
  const __bf16* Bh = &Bs_hi[col_tile * 512 + lane * 16];
  const __bf16* Bl = &Bs_lo[col_tile * 512 + lane * 16];

  for (int k0 = 0; k0 < K; k0 += 32) {
    // stage A tile: ROWS_PB x 32, split f32 -> bf16 hi + bf16 lo
    for (int idx = tid; idx < ROWS_PB * 32; idx += 256) {
      int r = idx >> 5, kk = idx & 31;
      int row = row0 + r;
      float v = (row < M) ? A[(size_t)row * K + k0 + kk] : 0.0f;
      __bf16 hi = (__bf16)v;
      As_hi[idx] = hi;
      As_lo[idx] = (__bf16)(v - (float)hi);
      // prefetch next k-chunk of the streaming A operand
      if (k0 + 32 < K && row < M && (kk & 15) == 0)
        __builtin_prefetch(&A[(size_t)row * K + k0 + 32 + kk], 0, 0);
    }
    // stage B tile swizzled to fragment order:
    //   lane = (kk>>4)*16 + (cc&15), slot = kk&15  (K = khalf*16 + slot)
    for (int idx = tid; idx < 32 * NCOLPAD; idx += 256) {
      int kk = idx / NCOLPAD, cc = idx % NCOLPAD;
      float v = (cc < ncol) ? B[(size_t)(k0 + kk) * ncol + cc] : 0.0f;
      __bf16 hi = (__bf16)v;
      int sw = (cc >> 4) * 512 + (((kk >> 4) << 4) + (cc & 15)) * 16 + (kk & 15);
      Bs_hi[sw] = hi;
      Bs_lo[sw] = (__bf16)(v - (float)hi);
    }
    __syncthreads();

    // A fragment (16x32): lane holds K {half*16..half*16+15}
    v16bf a_hi, a_lo, b_hi, b_lo;
#pragma unroll
    for (int e = 0; e < 8; ++e) {
      int kA = half * 16 + e;
      a_hi[e]     = Ah[kA];
      a_hi[8 + e] = Ah[8 + kA];
      a_lo[e]     = Al[kA];
      a_lo[8 + e] = Al[8 + kA];
    }
    // B fragment: contiguous 16 bf16 per lane (pre-swizzled)
#pragma unroll
    for (int e = 0; e < 16; ++e) {
      b_hi[e] = Bh[e];
      b_lo[e] = Bl[e];
    }
    // fp32-faithful product: hi*hi + hi*lo + lo*hi
    acc = __builtin_amdgcn_wmma_f32_16x16x32_bf16(false, a_hi, false, b_hi, (short)0, acc, false, false);
    acc = __builtin_amdgcn_wmma_f32_16x16x32_bf16(false, a_hi, false, b_lo, (short)0, acc, false, false);
    acc = __builtin_amdgcn_wmma_f32_16x16x32_bf16(false, a_lo, false, b_hi, (short)0, acc, false, false);
    __syncthreads();
  }

  // D layout: lanes 0-15: N=lane, M=i ; lanes 16-31: N=lane-16, M=8+i
  const int col = col_tile * 16 + mlow;
  if (col < ncol) {
#pragma unroll
    for (int i = 0; i < 8; ++i) {
      int row = row0 + row_tile * 16 + half * 8 + i;
      if (row < M) D[(size_t)row * ncol + col] = acc[i];
    }
  }
}

// ---- attention logits: a_src[n,h] = <h[n,h,:], att_src[h,:]> ---------------
__global__ void att_proj(const float* __restrict__ h, const float* __restrict__ att_s,
                         const float* __restrict__ att_d, float* __restrict__ as_,
                         float* __restrict__ ad_, int NH, int H, int C) {
  int i = blockIdx.x * blockDim.x + threadIdx.x;
  if (i >= NH) return;
  int hh = i % H;
  const float* hp = h + (size_t)i * C;
  float s = 0.f, d = 0.f;
  for (int c = 0; c < C; ++c) {
    float v = hp[c];
    s += v * att_s[hh * C + c];
    d += v * att_d[hh * C + c];
  }
  as_[i] = s;
  ad_[i] = d;
}

__device__ __forceinline__ void edge_sd(const int* ei, int e, int E, int N, int& s, int& d) {
  if (e < E) { s = ei[e]; d = ei[E + e]; }
  else       { s = d = e - E; }
  (void)N;
}

// ---- segment max over incoming edges ---------------------------------------
__global__ void edge_max(const int* __restrict__ ei, const float* __restrict__ as_,
                         const float* __restrict__ ad_, unsigned* __restrict__ amax,
                         int E, int N, int H) {
  int e = blockIdx.x * blockDim.x + threadIdx.x;
  if (e >= E + N) return;
  int s, d; edge_sd(ei, e, E, N, s, d);
  for (int h = 0; h < H; ++h) {
    float a = as_[(size_t)s * H + h] + ad_[(size_t)d * H + h];
    a = (a > 0.f) ? a : LRELU * a;
    atomicMax(&amax[(size_t)d * H + h], f2ord(a));
  }
}

// ---- segment sum of exp(alpha - amax) --------------------------------------
__global__ void edge_sum(const int* __restrict__ ei, const float* __restrict__ as_,
                         const float* __restrict__ ad_, const unsigned* __restrict__ amax,
                         float* __restrict__ asum, int E, int N, int H) {
  int e = blockIdx.x * blockDim.x + threadIdx.x;
  if (e >= E + N) return;
  int s, d; edge_sd(ei, e, E, N, s, d);
  for (int h = 0; h < H; ++h) {
    float a = as_[(size_t)s * H + h] + ad_[(size_t)d * H + h];
    a = (a > 0.f) ? a : LRELU * a;
    float w = expf(a - ord2f(amax[(size_t)d * H + h]));
    atomicAdd(&asum[(size_t)d * H + h], w);
  }
}

// ---- weighted scatter-add of messages: thread = (edge, head) ---------------
__global__ void edge_aggr(const int* __restrict__ ei, const float* __restrict__ as_,
                          const float* __restrict__ ad_, const unsigned* __restrict__ amax,
                          const float* __restrict__ asum, const float* __restrict__ hfeat,
                          float* __restrict__ agg, int E, int N, int H, int C) {
  int t = blockIdx.x * blockDim.x + threadIdx.x;
  int ET = E + N;
  if (t >= ET * H) return;
  int e = t / H, h = t % H;
  int s, d; edge_sd(ei, e, E, N, s, d);
  float a = as_[(size_t)s * H + h] + ad_[(size_t)d * H + h];
  a = (a > 0.f) ? a : LRELU * a;
  float w = expf(a - ord2f(amax[(size_t)d * H + h]));
  float coef = w / (asum[(size_t)d * H + h] + EPSD);
  const float* src = hfeat + ((size_t)s * H + h) * C;
  float* dst = agg + ((size_t)d * H + h) * C;
  for (int c = 0; c < C; ++c) atomicAdd(&dst[c], src[c] * coef);
}

// ---- bias + ELU, in place --------------------------------------------------
__global__ void bias_elu(float* __restrict__ v, const float* __restrict__ b, int F, size_t n) {
  size_t i = (size_t)blockIdx.x * blockDim.x + threadIdx.x;
  if (i >= n) return;
  float x = v[i] + b[i % F];
  v[i] = (x > 0.f) ? x : expm1f(x);
}

// ---- final: out = log_softmax(agg2 + b2), 19 classes -----------------------
__global__ void final_logsoftmax(const float* __restrict__ agg, const float* __restrict__ b,
                                 float* __restrict__ out, int N, int C) {
  int n = blockIdx.x * blockDim.x + threadIdx.x;
  if (n >= N) return;
  float v[32];
  float m = -3.402823466e38f;
  for (int c = 0; c < C; ++c) { v[c] = agg[(size_t)n * C + c] + b[c]; m = fmaxf(m, v[c]); }
  float s = 0.f;
  for (int c = 0; c < C; ++c) s += expf(v[c] - m);
  float lse = m + logf(s);
  for (int c = 0; c < C; ++c) out[(size_t)n * C + c] = v[c] - lse;
}

static inline int cdiv(long long a, long long b) { return (int)((a + b - 1) / b); }

extern "C" void kernel_launch(void* const* d_in, const int* in_sizes, int n_in,
                              void* d_out, int out_size, void* d_ws, size_t ws_size,
                              hipStream_t stream) {
  const int IN_CH = 1024, H1 = 16, C1 = 8, F1 = 128, C2 = 19;
  const int N = in_sizes[0] / IN_CH;
  const int E = in_sizes[1] / 2;
  const int ET = E + N;

  const float* x   = (const float*)d_in[0];
  const int*   ei  = (const int*)d_in[1];
  const float* W1  = (const float*)d_in[2];
  const float* aS1w= (const float*)d_in[3];
  const float* aD1w= (const float*)d_in[4];
  const float* b1  = (const float*)d_in[5];
  const float* W2  = (const float*)d_in[6];
  const float* aS2w= (const float*)d_in[7];
  const float* aD2w= (const float*)d_in[8];
  const float* b2  = (const float*)d_in[9];
  float* out = (float*)d_out;

  // ---- workspace layout (floats) ----
  float* ws   = (float*)d_ws;
  float* h1   = ws;                               // N*128
  float* agg1 = h1 + (size_t)N * F1;              // N*128 (becomes layer-2 input)
  float* aS1  = agg1 + (size_t)N * F1;            // N*16
  float* aD1  = aS1 + (size_t)N * H1;             // N*16
  unsigned* amax1 = (unsigned*)(aD1 + (size_t)N * H1);   // N*16
  float* asum1 = (float*)amax1 + (size_t)N * H1;  // N*16
  // layer 2 reuses h1 region (h1 is dead after aggregation)
  float* h2   = h1;                               // N*19
  float* aS2  = h1 + (size_t)N * C2;              // N
  float* aD2  = aS2 + N;                          // N
  unsigned* amax2 = (unsigned*)(aD2 + N);         // N
  float* asum2 = (float*)amax2 + N;               // N
  float* agg2  = asum2 + N;                       // N*19

  const int T = 256;
  dim3 blk(T);

  // ===================== Layer 1 =====================
  gemm_bf16x<8><<<cdiv(N, 16), blk, 0, stream>>>(x, W1, h1, N, IN_CH, F1);
  att_proj<<<cdiv((long long)N * H1, T), blk, 0, stream>>>(h1, aS1w, aD1w, aS1, aD1, N * H1, H1, C1);

  fill_u32<<<2048, blk, 0, stream>>>(amax1, ORD_NEG_INF, (size_t)N * H1);
  fill_u32<<<2048, blk, 0, stream>>>((unsigned*)asum1, 0u, (size_t)N * H1);
  fill_u32<<<2048, blk, 0, stream>>>((unsigned*)agg1, 0u, (size_t)N * F1);

  edge_max<<<cdiv(ET, T), blk, 0, stream>>>(ei, aS1, aD1, amax1, E, N, H1);
  edge_sum<<<cdiv(ET, T), blk, 0, stream>>>(ei, aS1, aD1, amax1, asum1, E, N, H1);
  edge_aggr<<<cdiv((long long)ET * H1, T), blk, 0, stream>>>(ei, aS1, aD1, amax1, asum1, h1, agg1, E, N, H1, C1);
  bias_elu<<<cdiv((long long)N * F1, T), blk, 0, stream>>>(agg1, b1, F1, (size_t)N * F1);

  // ===================== Layer 2 =====================
  gemm_bf16x<2><<<cdiv(N, 64), blk, 0, stream>>>(agg1, W2, h2, N, F1, C2);
  att_proj<<<cdiv(N, T), blk, 0, stream>>>(h2, aS2w, aD2w, aS2, aD2, N, 1, C2);

  fill_u32<<<2048, blk, 0, stream>>>(amax2, ORD_NEG_INF, (size_t)N);
  fill_u32<<<2048, blk, 0, stream>>>((unsigned*)asum2, 0u, (size_t)N);
  fill_u32<<<2048, blk, 0, stream>>>((unsigned*)agg2, 0u, (size_t)N * C2);

  edge_max<<<cdiv(ET, T), blk, 0, stream>>>(ei, aS2, aD2, amax2, E, N, 1);
  edge_sum<<<cdiv(ET, T), blk, 0, stream>>>(ei, aS2, aD2, amax2, asum2, E, N, 1);
  edge_aggr<<<cdiv(ET, T), blk, 0, stream>>>(ei, aS2, aD2, amax2, asum2, h2, agg2, E, N, 1, C2);

  final_logsoftmax<<<cdiv(N, T), blk, 0, stream>>>(agg2, b2, out, N, C2);
  (void)n_in; (void)out_size; (void)ws_size;
}
